// CrossMultiheadDiffAttn_67319317397795
// MI455X (gfx1250) — matrix operand order repaired and checked by reading, not compiled, not run on hardware
//
#include <hip/hip_runtime.h>
#include <hip/hip_bf16.h>

// ---------------- problem constants ----------------
#define EMBED   1024
#define HEADS   16
#define HEAD_DIM 32
#define BATCH   2
#define SEQT    2048
#define SEQS    2048
#define BT      (BATCH * SEQT)      // 4096 rows for all GEMMs
constexpr float LAMBDA_INIT = 0.7836057665f;   // 0.8 - 0.6*exp(-0.3*12)
constexpr float SCALING     = 0.17677669529663687f; // 32^-0.5

typedef __attribute__((ext_vector_type(16))) _Float16 v16h;
typedef __attribute__((ext_vector_type(8)))  _Float16 v8h;
typedef __attribute__((ext_vector_type(8)))  float    v8f;

// ---------------- CDNA5 async copy / wait helpers ----------------
__device__ inline void async_copy_b128(unsigned lds_off, const void* gptr) {
  unsigned long long ga = (unsigned long long)gptr;
  asm volatile("global_load_async_to_lds_b128 %0, %1, off"
               :: "v"(lds_off), "v"(ga) : "memory");
}
__device__ inline void wait_async0() { asm volatile("s_wait_asynccnt 0x0" ::: "memory"); }
__device__ inline void wait_async1() { asm volatile("s_wait_asynccnt 0x1" ::: "memory"); }
__device__ inline void wait_ds0()    { asm volatile("s_wait_dscnt 0x0"    ::: "memory"); }

// ---------------- WMMA fragment loaders (ISA 7.12.2, wave32), 16B-vectorized ----------------
// A 16x32 f16: lane m=lane%16, g=lane/16 reads row m: k in [8g,8g+8) and [16+8g,16+8g+8).
__device__ inline v16h load_a_f16(const _Float16* __restrict__ src, int ld) {
  int lane = threadIdx.x & 31;
  int m = lane & 15, g = lane >> 4;
  const _Float16* r = src + m * ld + 8 * g;
  v8h lo = *reinterpret_cast<const v8h*>(r);
  v8h hi = *reinterpret_cast<const v8h*>(r + 16);
  v16h a;
#pragma unroll
  for (int i = 0; i < 8; ++i) { a[i] = lo[i]; a[8 + i] = hi[i]; }
  return a;
}

// B 32x16 f16 where memory holds B^T (K-contiguous): element (k,n) at src[n*ld + k].
// lane n=lane%16, g=lane/16 reads k in [8g,8g+8) and [16+8g,16+8g+8) -> 2 x b128.
__device__ inline v16h load_bT_f16(const _Float16* __restrict__ src, int ld) {
  int lane = threadIdx.x & 31;
  int n = lane & 15, g = lane >> 4;
  const _Float16* r = src + n * ld + 8 * g;
  v8h lo = *reinterpret_cast<const v8h*>(r);
  v8h hi = *reinterpret_cast<const v8h*>(r + 16);
  v16h b;
#pragma unroll
  for (int i = 0; i < 8; ++i) { b[i] = lo[i]; b[8 + i] = hi[i]; }
  return b;
}

// ---------------- fp32 -> f16 bulk conversion ----------------
__global__ __launch_bounds__(256) void cvt_f32_f16(const float* __restrict__ in,
                                                   _Float16* __restrict__ out) {
  size_t i = ((size_t)blockIdx.x * 256 + threadIdx.x) * 4;
  float4 v = *reinterpret_cast<const float4*>(in + i);
  out[i + 0] = (_Float16)v.x;
  out[i + 1] = (_Float16)v.y;
  out[i + 2] = (_Float16)v.z;
  out[i + 3] = (_Float16)v.w;
}

// ---------------- fp32 [rows][cols] -> f16 [cols][rows] (weights) ----------------
__global__ __launch_bounds__(256) void cvt_transpose_f32_f16(const float* __restrict__ in,
                                                             _Float16* __restrict__ out,
                                                             int rows, int cols) {
  __shared__ _Float16 t[64][72];
  const int bx = blockIdx.x * 64;   // col base
  const int by = blockIdx.y * 64;   // row base
  const int tid = threadIdx.x;
  {
    int lr = tid >> 4, lc = (tid & 15) * 4;
#pragma unroll
    for (int p = 0; p < 4; ++p) {
      int r = lr + p * 16;
      float4 v = *reinterpret_cast<const float4*>(in + (size_t)(by + r) * cols + bx + lc);
      t[lc + 0][r] = (_Float16)v.x;
      t[lc + 1][r] = (_Float16)v.y;
      t[lc + 2][r] = (_Float16)v.z;
      t[lc + 3][r] = (_Float16)v.w;
    }
  }
  __syncthreads();
  {
    int lr = tid >> 3, lc = (tid & 7) * 8;
#pragma unroll
    for (int p = 0; p < 2; ++p) {
      int c = lr + p * 32;
      v8h v;
#pragma unroll
      for (int i = 0; i < 8; ++i) v[i] = t[c][lc + i];
      *reinterpret_cast<v8h*>(out + (size_t)(bx + c) * rows + by + lc) = v;
    }
  }
}

// ---------------- f16 [rows][cols] -> f16 [cols][rows], batched over z ----------------
__global__ __launch_bounds__(256) void transpose_f16(const _Float16* __restrict__ in,
                                                     _Float16* __restrict__ out,
                                                     int rows, int cols) {
  __shared__ _Float16 t[64][72];
  const size_t mat = (size_t)rows * cols;
  const _Float16* inb = in + (size_t)blockIdx.z * mat;
  _Float16* outb = out + (size_t)blockIdx.z * mat;
  const int bx = blockIdx.x * 64, by = blockIdx.y * 64, tid = threadIdx.x;
  const int lr = tid >> 3, lc = (tid & 7) * 8;
#pragma unroll
  for (int p = 0; p < 2; ++p) {
    int r = lr + p * 32;
    v8h v = *reinterpret_cast<const v8h*>(inb + (size_t)(by + r) * cols + bx + lc);
#pragma unroll
    for (int i = 0; i < 8; ++i) t[lc + i][r] = v[i];
  }
  __syncthreads();
#pragma unroll
  for (int p = 0; p < 2; ++p) {
    int c = lr + p * 32;
    v8h v;
#pragma unroll
    for (int i = 0; i < 8; ++i) v[i] = t[c][lc + i];
    *reinterpret_cast<v8h*>(outb + (size_t)(bx + c) * rows + by + lc) = v;
  }
}

// ---------------- lambda scalar ----------------
__global__ void lambda_kernel(const float* __restrict__ lq1, const float* __restrict__ lk1,
                              const float* __restrict__ lq2, const float* __restrict__ lk2,
                              float* __restrict__ out) {
  int lane = threadIdx.x;  // 32 threads, HEAD_DIM == 32
  float p1 = lq1[lane] * lk1[lane];
  float p2 = lq2[lane] * lk2[lane];
#pragma unroll
  for (int off = 1; off < 32; off <<= 1) {
    p1 += __shfl_xor(p1, off, 32);
    p2 += __shfl_xor(p2, off, 32);
  }
  if (lane == 0) out[0] = __expf(p1) - __expf(p2) + LAMBDA_INIT;
}

// ---------------- f16 WMMA GEMM ----------------
// C[M,N] = (A[M,K] @ B[K,N]) * oscale, where B is passed TRANSPOSED (Bt = [N][K], L2-resident).
// A tile (64x32, shared by 8 waves) staged via async-to-LDS double buffer; B streams from L2.
template <typename OT>
__global__ __launch_bounds__(256) void gemm_wmma_f16(const _Float16* __restrict__ A,
                                                     const _Float16* __restrict__ Bt,
                                                     OT* __restrict__ C,
                                                     int M, int N, int K, float oscale) {
  constexpr int BM = 64, BN = 128, BK = 32;
  __shared__ __align__(16) _Float16 As[2][BM * BK];   // 2 x 4 KB
  const int tid  = threadIdx.x;
  const int wave = tid >> 5;
  const int wr   = wave >> 2;           // 0..1 -> 32-row strip
  const int wc   = wave & 3;            // 0..3 -> 32-col strip
  const int rowBlk = blockIdx.y * BM;
  const int colBlk = blockIdx.x * BN;

  // one 16-byte async chunk per thread per A tile
  const int ar = tid >> 2, ac = (tid & 3) * 8;
  auto issue_tile = [&](int buf, int k0) {
    async_copy_b128((unsigned)(uintptr_t)&As[buf][ar * BK + ac],
                    A + (size_t)(rowBlk + ar) * K + k0 + ac);
  };

  v8f acc[2][2] = {};
  issue_tile(0, 0);
  for (int k0 = 0; k0 < K; k0 += BK) {
    const int buf = (k0 / BK) & 1;
    if (k0 + BK < K) { issue_tile(buf ^ 1, k0 + BK); wait_async1(); }
    else             { wait_async0(); }
    __syncthreads();

    v16h bf0 = load_bT_f16(Bt + (size_t)(colBlk + wc * 32)      * K + k0, K);
    v16h bf1 = load_bT_f16(Bt + (size_t)(colBlk + wc * 32 + 16) * K + k0, K);
#pragma unroll
    for (int i = 0; i < 2; ++i) {
      v16h a = load_a_f16(&As[buf][(wr * 32 + i * 16) * BK], BK);
      acc[i][0] = __builtin_amdgcn_wmma_f32_16x16x32_f16(false, a, false, bf0,
                                                         (short)0, acc[i][0], false, false);
      acc[i][1] = __builtin_amdgcn_wmma_f32_16x16x32_f16(false, a, false, bf1,
                                                         (short)0, acc[i][1], false, false);
    }
    __syncthreads();
  }

  // epilogue: C layout m = v + 8*(lane/16), n = lane%16
  const int lane = tid & 31;
  const int n = lane & 15, g = lane >> 4;
#pragma unroll
  for (int i = 0; i < 2; ++i)
#pragma unroll
    for (int j = 0; j < 2; ++j)
#pragma unroll
      for (int v = 0; v < 8; ++v) {
        int row = rowBlk + wr * 32 + i * 16 + v + 8 * g;
        int col = colBlk + wc * 32 + j * 16 + n;
        C[(size_t)row * N + col] = (OT)(acc[i][j][v] * oscale);
      }
}

// ---------------- fused differential attention ----------------
// 8 waves per block, all sharing one (batch, head-pair); wave w owns query rows
// [t0 + 16w, t0 + 16w + 16). All waves issue identical K/V fragment addresses per
// S-step, so the WGP cache collapses them (8x less L2 traffic than 1-wave blocks).
// No cross-wave barriers needed: each wave has a private P-staging LDS region,
// ordered by the in-order DS pipe + explicit s_wait_dscnt. No-max softmax
// (scores ~N(0,1) after the folded D^-1/2 scale -> exp is fp32-safe).
__global__ __launch_bounds__(256) void diff_attn(const _Float16* __restrict__ Qh,
                                                 const _Float16* __restrict__ Kh,
                                                 const _Float16* __restrict__ Vt,
                                                 const float* __restrict__ lam_ptr,
                                                 const float* __restrict__ subw,
                                                 _Float16* __restrict__ AO) {
  __shared__ __align__(16) _Float16 Pb[8][16 * 64];  // 16 KB, wave-private regions
  const int wave = threadIdx.x >> 5;
  const int lane = threadIdx.x & 31;
  const int n = lane & 15, g = lane >> 4;
  const int b = blockIdx.z, h = blockIdx.y;
  const int t0 = blockIdx.x * 128 + wave * 16;
  const float lambda_full = *lam_ptr;

  v8f ocomb[4] = {};

  for (int half = 0; half < 2; ++half) {
    const int hq = 2 * h + half;
    const v16h aQ = load_a_f16(Qh + (size_t)(b * SEQT + t0) * EMBED + hq * HEAD_DIM, EMBED);

    float rs[8];
#pragma unroll
    for (int v = 0; v < 8; ++v) rs[v] = 0.0f;
    v8f o[4] = {};

    for (int s0 = 0; s0 < SEQS; s0 += 64) {
      // scores 16x64 (Q already carries D^-1/2)
      v8f s[4];
#pragma unroll
      for (int j = 0; j < 4; ++j) {
        v16h bK = load_bT_f16(Kh + (size_t)(b * SEQS + s0 + j * 16) * EMBED + hq * HEAD_DIM,
                              EMBED);
        v8f z = {};
        s[j] = __builtin_amdgcn_wmma_f32_16x16x32_f16(false, aQ, false, bK,
                                                      (short)0, z, false, false);
      }
#pragma unroll
      for (int j = 0; j < 4; ++j)
#pragma unroll
        for (int v = 0; v < 8; ++v) {
          float e = __expf(s[j][v]);
          s[j][v] = e;
          rs[v] += e;
        }

      // P (C-layout) -> LDS -> A-layout fragments; wave-private, DS in-order
#pragma unroll
      for (int v = 0; v < 8; ++v)
#pragma unroll
        for (int j = 0; j < 4; ++j)
          Pb[wave][(v + 8 * g) * 64 + j * 16 + n] = (_Float16)s[j][v];
      wait_ds0();

#pragma unroll
      for (int j2 = 0; j2 < 2; ++j2) {
        v16h aP = load_a_f16(&Pb[wave][j2 * 32], 64);
#pragma unroll
        for (int e = 0; e < 4; ++e) {
          v16h bV = load_bT_f16(Vt + ((size_t)b * EMBED + h * 64 + e * 16) * SEQS
                                     + s0 + j2 * 32, SEQS);
          o[e] = __builtin_amdgcn_wmma_f32_16x16x32_f16(false, aP, false, bV,
                                                        (short)0, o[e], false, false);
        }
      }
    }

    // one shuffle-reduce per pass for softmax denominators, then fold in
    const float w = (half == 0) ? 1.0f : -lambda_full;
#pragma unroll
    for (int v = 0; v < 8; ++v) {
      float t = rs[v];
#pragma unroll
      for (int off = 1; off < 16; off <<= 1) t += __shfl_xor(t, off, 32);
      float inv = w / t;
#pragma unroll
      for (int e = 0; e < 4; ++e) ocomb[e][v] += o[e][v] * inv;
    }
  }

  // fused RMS-norm over 64 cols + (1 - lambda_init) scale, store f16
#pragma unroll
  for (int v = 0; v < 8; ++v) {
    float ss = 0.0f;
#pragma unroll
    for (int e = 0; e < 4; ++e) ss += ocomb[e][v] * ocomb[e][v];
#pragma unroll
    for (int off = 1; off < 16; off <<= 1) ss += __shfl_xor(ss, off, 32);
    float r = rsqrtf(ss * (1.0f / 64.0f) + 1e-5f) * (1.0f - LAMBDA_INIT);
    int trow = t0 + v + 8 * g;
#pragma unroll
    for (int e = 0; e < 4; ++e) {
      int col = e * 16 + n;
      float val = ocomb[e][v] * r * subw[col];
      AO[(size_t)(b * SEQT + trow) * EMBED + h * 64 + col] = (_Float16)val;
    }
  }
}

// ---------------- launch ----------------
extern "C" void kernel_launch(void* const* d_in, const int* in_sizes, int n_in,
                              void* d_out, int out_size, void* d_ws, size_t ws_size,
                              hipStream_t stream) {
  const float* query_x = (const float*)d_in[0];
  const float* kv_x    = (const float*)d_in[1];
  const float* Wq      = (const float*)d_in[2];
  const float* Wk      = (const float*)d_in[3];
  const float* Wv      = (const float*)d_in[4];
  const float* Wo      = (const float*)d_in[5];
  const float* lq1     = (const float*)d_in[6];
  const float* lk1     = (const float*)d_in[7];
  const float* lq2     = (const float*)d_in[8];
  const float* lk2     = (const float*)d_in[9];
  const float* subln_w = (const float*)d_in[10];
  float* out = (float*)d_out;

  // workspace layout (f16 elements)
  const size_t act  = (size_t)BT * EMBED;      // 4M elements
  const size_t wmat = (size_t)EMBED * EMBED;   // 1M elements
  char* ws = (char*)d_ws;
  float*    lam = (float*)ws;
  _Float16* Xq  = (_Float16*)(ws + 256);
  _Float16* Xkv = Xq  + act;
  _Float16* WqT = Xkv + act;
  _Float16* WkT = WqT + wmat;
  _Float16* WvT = WkT + wmat;
  _Float16* WoT = WvT + wmat;
  _Float16* Qh  = WoT + wmat;
  _Float16* Kh  = Qh  + act;
  _Float16* Vh  = Kh  + act;
  _Float16* Vt  = Vh  + act;
  _Float16* AOh = Vt  + act;

  lambda_kernel<<<1, 32, 0, stream>>>(lq1, lk1, lq2, lk2, lam);

  // one-time f16 conversions; weights converted AND transposed -> K-contiguous B operands
  cvt_f32_f16<<<act / 1024, 256, 0, stream>>>(query_x, Xq);
  cvt_f32_f16<<<act / 1024, 256, 0, stream>>>(kv_x,    Xkv);
  dim3 wtgrid(EMBED / 64, EMBED / 64);
  cvt_transpose_f32_f16<<<wtgrid, 256, 0, stream>>>(Wq, WqT, EMBED, EMBED);
  cvt_transpose_f32_f16<<<wtgrid, 256, 0, stream>>>(Wk, WkT, EMBED, EMBED);
  cvt_transpose_f32_f16<<<wtgrid, 256, 0, stream>>>(Wv, WvT, EMBED, EMBED);
  cvt_transpose_f32_f16<<<wtgrid, 256, 0, stream>>>(Wo, WoT, EMBED, EMBED);

  dim3 ggrid(EMBED / 128, BT / 64);   // (8, 64)
  gemm_wmma_f16<_Float16><<<ggrid, 256, 0, stream>>>(Xq,  WqT, Qh, BT, EMBED, EMBED, SCALING);
  gemm_wmma_f16<_Float16><<<ggrid, 256, 0, stream>>>(Xkv, WkT, Kh, BT, EMBED, EMBED, 1.0f);
  gemm_wmma_f16<_Float16><<<ggrid, 256, 0, stream>>>(Xkv, WvT, Vh, BT, EMBED, EMBED, 1.0f);

  // V -> V^T per batch so P@V streams K-contiguous b128 fragments
  transpose_f16<<<dim3(EMBED / 64, SEQS / 64, BATCH), 256, 0, stream>>>(Vh, Vt, SEQS, EMBED);

  diff_attn<<<dim3(SEQT / 128, HEADS, BATCH), 256, 0, stream>>>(Qh, Kh, Vt, lam, subln_w, AOh);

  gemm_wmma_f16<float><<<ggrid, 256, 0, stream>>>(AOh, WoT, out, BT, EMBED, EMBED, 1.0f);

  (void)in_sizes; (void)n_in; (void)out_size; (void)ws_size;
}
